// FlattenIntraCycleMoELayer_51616916963603
// MI455X (gfx1250) — matrix operand order repaired and checked by reading, not compiled, not run on hardware
//
#include <hip/hip_runtime.h>
#include <cstdint>
#include <cstddef>

// ---------------------------------------------------------------------------
// Types for CDNA5 WMMA (wave32): v_wmma_f32_16x16x32_bf16
// ---------------------------------------------------------------------------
typedef __attribute__((ext_vector_type(16))) __bf16        v16bf;
typedef __attribute__((ext_vector_type(8)))  float         v8f;
typedef __attribute__((ext_vector_type(4)))  unsigned int  u32x4;
typedef __attribute__((ext_vector_type(2)))  unsigned int  u32x2;

union FragBF { v16bf v; u32x4 q[2]; };
union F4     { float4 v; float a[4]; };

// f32 pair -> packed bf16 {lo, hi}: round-half-up bias then one v_perm_b32
// (result bytes, LSB first: lo[2], lo[3], hi[2], hi[3] -> sel 0x07060302 with
//  perm(src0=hi, src1=lo); byte idx 0-3 select src1, 4-7 select src0)
__device__ __forceinline__ unsigned int pk_bf16(float lo, float hi) {
  union { float f; unsigned int u; } a, b;
  a.f = lo; b.f = hi;
  return __builtin_amdgcn_perm(b.u + 0x8000u, a.u + 0x8000u, 0x07060302u);
}

__device__ __forceinline__ float gelu_tanh(float x) {
  const float c0 = 0.7978845608028654f;   // sqrt(2/pi)
  const float c1 = 0.044715f;
  float t = tanhf(c0 * (x + c1 * x * x * x));
  return 0.5f * x * (1.0f + t);
}

// Problem constants
#define BATT   64
#define LCYC   512
#define KDIM   900        // C*F = 3*300
#define DMODEL 512
#define DLLM   4096
#define DFF    2048
#define NEXP   8
#define SA     40         // LDS k-stride (bf16 elems), padded from 32

// ---------------------------------------------------------------------------
// Kernel 1: h = gelu([DKP ; cyc] @ W1 + b1)  — WMMA bf16, K = 4096, double-
// buffered LDS with register prefetch; rank-1 cyc*W1[4096,:] in epilogue.
// Grid: 16 blocks (128 cols each), 256 threads (8 waves; wave w -> 16 cols).
// ---------------------------------------------------------------------------
__global__ __launch_bounds__(256) void router_h_kernel(
    const float* __restrict__ dkp, const float* __restrict__ cyc,
    const float* __restrict__ W1,  const float* __restrict__ b1,
    float* __restrict__ h)
{
  const int c0   = blockIdx.x * 128;
  const int tid  = threadIdx.x;
  const int lane = tid & 31, wv = tid >> 5;
  const int l16  = lane & 15, lh = lane >> 4;

  __shared__ alignas(16) unsigned short As[2 * 64 * SA];
  __shared__ alignas(16) unsigned short Bs[2 * 128 * SA];

  const v8f vz = {0.f, 0.f, 0.f, 0.f, 0.f, 0.f, 0.f, 0.f};
  v8f acc[4];
#pragma unroll
  for (int m = 0; m < 4; ++m) acc[m] = vz;

  // fragment offsets (per lane, fixed)
  int aoff[4];
#pragma unroll
  for (int m = 0; m < 4; ++m) aoff[m] = (m * 16 + l16) * SA + lh * 16;
  const int boff = (wv * 16 + l16) * SA + lh * 16;

  F4 pa[2];          // A prefetch
  F4 pb[2][2];       // B prefetch: [iter][k-row]

  auto prefetch = [&](int ks) {
    const int k0 = ks * 32;
#pragma unroll
    for (int i = 0; i < 2; ++i) {
      int idx = tid + i * 256;
      int row = idx >> 3, kq = idx & 7;
      pa[i].v = *(const float4*)(dkp + (size_t)row * DLLM + k0 + kq * 4);
    }
#pragma unroll
    for (int i = 0; i < 2; ++i) {
      int idx = tid + i * 256;
      int kk2 = idx >> 5;
      int cg  = c0 + (idx & 31) * 4;
      int kg  = k0 + kk2 * 2;
      pb[i][0].v = *(const float4*)(W1 + (size_t)kg * DFF + cg);
      pb[i][1].v = *(const float4*)(W1 + (size_t)(kg + 1) * DFF + cg);
    }
  };

  auto commit = [&](int buf) {
    unsigned short* Ab = As + buf * 64 * SA;
    unsigned short* Bb = Bs + buf * 128 * SA;
#pragma unroll
    for (int i = 0; i < 2; ++i) {
      int idx = tid + i * 256;
      int row = idx >> 3, kq = idx & 7;
      u32x2 st;
      st.x = pk_bf16(pa[i].a[0], pa[i].a[1]);
      st.y = pk_bf16(pa[i].a[2], pa[i].a[3]);
      *(u32x2*)&Ab[row * SA + kq * 4] = st;
    }
#pragma unroll
    for (int i = 0; i < 2; ++i) {
      int idx = tid + i * 256;
      int kk2 = idx >> 5, cq = idx & 31;
#pragma unroll
      for (int j = 0; j < 4; ++j)
        *(unsigned int*)&Bb[(cq * 4 + j) * SA + kk2 * 2] =
            pk_bf16(pb[i][0].a[j], pb[i][1].a[j]);
    }
  };

  auto compute = [&](int buf) {
    const unsigned short* Ab = As + buf * 64 * SA;
    const unsigned short* Bb = Bs + buf * 128 * SA;
    FragBF bfr;
    bfr.q[0] = *(const u32x4*)&Bb[boff];
    bfr.q[1] = *(const u32x4*)&Bb[boff + 8];
#pragma unroll
    for (int m = 0; m < 4; ++m) {
      FragBF af;
      af.q[0] = *(const u32x4*)&Ab[aoff[m]];
      af.q[1] = *(const u32x4*)&Ab[aoff[m] + 8];
      acc[m] = __builtin_amdgcn_wmma_f32_16x16x32_bf16(
          false, af.v, false, bfr.v, (short)0, acc[m], false, false);
    }
  };

  const int NK = DLLM / 32;   // 128
  prefetch(0);
  commit(0);
  __syncthreads();
  for (int ks = 0; ks < NK - 1; ++ks) {
    prefetch(ks + 1);         // global loads overlap the WMMA block below
    compute(ks & 1);
    commit((ks + 1) & 1);     // writes other buffer; one barrier per step
    __syncthreads();
  }
  compute((NK - 1) & 1);

  // Epilogue: + cyc[row] * W1[4096, col] + b1[col], gelu
  const int   col   = c0 + wv * 16 + l16;
  const float wlast = W1[(size_t)DLLM * DFF + col];
  const float bias  = b1[col];
#pragma unroll
  for (int m = 0; m < 4; ++m) {
#pragma unroll
    for (int i = 0; i < 8; ++i) {
      int row = m * 16 + i + lh * 8;       // C/D layout: vgpr i, lane-half lh
      float v = acc[m][i] + cyc[row] * wlast + bias;
      h[(size_t)row * DFF + col] = gelu_tanh(v);
    }
  }
}

// ---------------------------------------------------------------------------
// Kernel 2: logits = h @ W2 + b2; top-2 softmax renormalized gates.
// One block per battery.
// ---------------------------------------------------------------------------
__global__ __launch_bounds__(128) void router_gates_kernel(
    const float* __restrict__ h, const float* __restrict__ W2,
    const float* __restrict__ b2, float* __restrict__ g_ws,
    int* __restrict__ i_ws)
{
  const int b = blockIdx.x, tid = threadIdx.x;
  float s[NEXP];
#pragma unroll
  for (int e = 0; e < NEXP; ++e) s[e] = 0.f;
  for (int k = tid; k < DFF; k += 128) {
    float hv = h[(size_t)b * DFF + k];
    const float* wr = W2 + (size_t)k * NEXP;
#pragma unroll
    for (int e = 0; e < NEXP; ++e) s[e] += hv * wr[e];
  }
  __shared__ float red[NEXP][128];
#pragma unroll
  for (int e = 0; e < NEXP; ++e) red[e][tid] = s[e];
  __syncthreads();
  __shared__ float logits[NEXP];
  if (tid < NEXP) {
    float a = 0.f;
    for (int i = 0; i < 128; ++i) a += red[tid][i];
    logits[tid] = a + b2[tid];
  }
  __syncthreads();
  if (tid == 0) {
    int i0 = 0;
    for (int e = 1; e < NEXP; ++e) if (logits[e] > logits[i0]) i0 = e;
    int i1 = (i0 == 0) ? 1 : 0;
    for (int e = 0; e < NEXP; ++e)
      if (e != i0 && logits[e] > logits[i1]) i1 = e;
    float mx = logits[0];
    for (int e = 1; e < NEXP; ++e) mx = fmaxf(mx, logits[e]);
    float Z = 0.f;
    for (int e = 0; e < NEXP; ++e) Z += __expf(logits[e] - mx);
    float p0 = __expf(logits[i0] - mx) / Z;
    float p1 = __expf(logits[i1] - mx) / Z;
    float d  = p0 + p1 + 1e-9f;
    g_ws[b * 2]     = p0 / d;
    g_ws[b * 2 + 1] = p1 / d;
    i_ws[b * 2]     = i0;
    i_ws[b * 2 + 1] = i1;
  }
}

// ---------------------------------------------------------------------------
// Kernel 3: fused general + MoE GEMM (per-battery combined weight).
//   out[b] = x[b] @ (gen_W + g0*eW[e0] + g1*eW[e1]) + combined bias
// Grid (4 col-tiles, 4 row-tiles, 64 batteries); 256 threads = 8 waves in a
// 4(M) x 2(N) wave grid; each wave: 32x64 output = 2 A-frags x 4 B-frags.
// K = 900 -> 29 steps of 32, last (partial) step peeled; double-buffered LDS
// with register prefetch so global latency overlaps the WMMA block.
// ---------------------------------------------------------------------------
__global__ __launch_bounds__(256) void moe_fused_gemm_kernel(
    const float* __restrict__ x,    const float* __restrict__ genW,
    const float* __restrict__ genB, const float* __restrict__ eW,
    const float* __restrict__ eB,   const float* __restrict__ g_ws,
    const int* __restrict__ i_ws,   float* __restrict__ out)
{
  const int b    = blockIdx.z;
  const int c0   = blockIdx.x * 128;
  const int l0   = blockIdx.y * 128;
  const int tid  = threadIdx.x;
  const int lane = tid & 31, wv = tid >> 5;
  const int wm   = wv >> 1,  wn = wv & 1;
  const int l16  = lane & 15, lh = lane >> 4;

  __shared__ alignas(16) unsigned short As[2 * 128 * SA];
  __shared__ alignas(16) unsigned short Bs[2 * 128 * SA];

  const float g0 = g_ws[b * 2], g1 = g_ws[b * 2 + 1];
  const int   e0 = i_ws[b * 2], e1 = i_ws[b * 2 + 1];
  const float* eW0 = eW + (size_t)e0 * KDIM * DMODEL;
  const float* eW1 = eW + (size_t)e1 * KDIM * DMODEL;
  const float* xb  = x  + (size_t)b * LCYC * KDIM;

  const v8f vz = {0.f, 0.f, 0.f, 0.f, 0.f, 0.f, 0.f, 0.f};
  v8f acc[2][4];
#pragma unroll
  for (int m = 0; m < 2; ++m)
#pragma unroll
    for (int n = 0; n < 4; ++n) acc[m][n] = vz;

  int aoff[2], boff[4];
#pragma unroll
  for (int m = 0; m < 2; ++m) aoff[m] = (wm * 32 + m * 16 + l16) * SA + lh * 16;
#pragma unroll
  for (int n = 0; n < 4; ++n) boff[n] = (wn * 64 + n * 16 + l16) * SA + lh * 16;

  F4 pa[4];                            // A prefetch: 4 float4
  F4 pg[2][2], p0r[2][2], p1r[2][2];   // B prefetch: [iter][k-row] x {gen,e0,e1}

  // guard==true: zero-fill chunks with k >= 900 (used only for the last step)
  auto prefetch = [&](int ks, bool guard) {
    const int k0 = ks * 32;
#pragma unroll
    for (int i = 0; i < 4; ++i) {
      int idx = tid + i * 256;
      int row = idx >> 3, kq = idx & 7;
      int kg = k0 + kq * 4;
      if (guard && kg >= KDIM) {
        pa[i].a[0] = pa[i].a[1] = pa[i].a[2] = pa[i].a[3] = 0.f;
      } else {
        pa[i].v = *(const float4*)(xb + (size_t)(l0 + row) * KDIM + kg);
      }
    }
#pragma unroll
    for (int i = 0; i < 2; ++i) {
      int idx = tid + i * 256;
      int kk2 = idx >> 5;
      int cg  = c0 + (idx & 31) * 4;
      int kg  = k0 + kk2 * 2;
      if (guard && kg >= KDIM) {   // kg even => kg < 900 implies kg+1 < 900
#pragma unroll
        for (int r = 0; r < 2; ++r)
#pragma unroll
          for (int j = 0; j < 4; ++j) {
            pg[i][r].a[j] = 0.f; p0r[i][r].a[j] = 0.f; p1r[i][r].a[j] = 0.f;
          }
      } else {
#pragma unroll
        for (int r = 0; r < 2; ++r) {
          pg[i][r].v  = *(const float4*)(genW + (size_t)(kg + r) * DMODEL + cg);
          p0r[i][r].v = *(const float4*)(eW0  + (size_t)(kg + r) * DMODEL + cg);
          p1r[i][r].v = *(const float4*)(eW1  + (size_t)(kg + r) * DMODEL + cg);
        }
      }
    }
  };

  auto commit = [&](int buf) {
    unsigned short* Ab = As + buf * 128 * SA;
    unsigned short* Bb = Bs + buf * 128 * SA;
#pragma unroll
    for (int i = 0; i < 4; ++i) {
      int idx = tid + i * 256;
      int row = idx >> 3, kq = idx & 7;
      u32x2 st;
      st.x = pk_bf16(pa[i].a[0], pa[i].a[1]);
      st.y = pk_bf16(pa[i].a[2], pa[i].a[3]);
      *(u32x2*)&Ab[row * SA + kq * 4] = st;
    }
#pragma unroll
    for (int i = 0; i < 2; ++i) {
      int idx = tid + i * 256;
      int kk2 = idx >> 5, cq = idx & 31;
      float w0[4], w1[4];
#pragma unroll
      for (int j = 0; j < 4; ++j) {
        w0[j] = pg[i][0].a[j] + g0 * p0r[i][0].a[j] + g1 * p1r[i][0].a[j];
        w1[j] = pg[i][1].a[j] + g0 * p0r[i][1].a[j] + g1 * p1r[i][1].a[j];
      }
#pragma unroll
      for (int j = 0; j < 4; ++j)
        *(unsigned int*)&Bb[(cq * 4 + j) * SA + kk2 * 2] = pk_bf16(w0[j], w1[j]);
    }
  };

  auto compute = [&](int buf) {
    const unsigned short* Ab = As + buf * 128 * SA;
    const unsigned short* Bb = Bs + buf * 128 * SA;
    FragBF af[2], bf[4];
#pragma unroll
    for (int m = 0; m < 2; ++m) {
      af[m].q[0] = *(const u32x4*)&Ab[aoff[m]];
      af[m].q[1] = *(const u32x4*)&Ab[aoff[m] + 8];
    }
#pragma unroll
    for (int n = 0; n < 4; ++n) {
      bf[n].q[0] = *(const u32x4*)&Bb[boff[n]];
      bf[n].q[1] = *(const u32x4*)&Bb[boff[n] + 8];
    }
#pragma unroll
    for (int m = 0; m < 2; ++m)
#pragma unroll
      for (int n = 0; n < 4; ++n)
        acc[m][n] = __builtin_amdgcn_wmma_f32_16x16x32_bf16(
            false, af[m].v, false, bf[n].v, (short)0, acc[m][n], false, false);
  };

  const int NK = 29;          // ceil(900/32); step 28 is partial (k 896..899)
  prefetch(0, false);
  commit(0);
  __syncthreads();
  for (int ks = 0; ks < NK - 1; ++ks) {
    prefetch(ks + 1, ks + 1 == NK - 1);  // globals overlap the WMMA block
    compute(ks & 1);
    commit((ks + 1) & 1);
    __syncthreads();
  }
  compute((NK - 1) & 1);

  // Epilogue: combined bias, write f32 output
  float biasn[4];
#pragma unroll
  for (int n = 0; n < 4; ++n) {
    int col = c0 + wn * 64 + n * 16 + l16;
    biasn[n] = genB[col] + g0 * eB[(size_t)e0 * DMODEL + col]
                         + g1 * eB[(size_t)e1 * DMODEL + col];
  }
#pragma unroll
  for (int m = 0; m < 2; ++m)
#pragma unroll
    for (int n = 0; n < 4; ++n)
#pragma unroll
      for (int i = 0; i < 8; ++i) {
        int row = l0 + wm * 32 + m * 16 + i + lh * 8;
        int col = c0 + wn * 64 + n * 16 + l16;
        out[((size_t)b * LCYC + row) * DMODEL + col] = acc[m][n][i] + biasn[n];
      }
}

// ---------------------------------------------------------------------------
// Host launch
// Inputs (setup_inputs order):
//  0 cycle_curve_data [64,512,3,300] f32
//  1 cycle_numbers    [64,1]         f32
//  2 DKP_embeddings   [64,4096]      f32
//  3 gate_W1 [4097,2048]  4 gate_b1 [2048]
//  5 gate_W2 [2048,8]     6 gate_b2 [8]
//  7 expert_W [8,900,512] 8 expert_b [8,512]
//  9 gen_W [900,512]     10 gen_b [512]
// Output: [64,512,512] f32
// ---------------------------------------------------------------------------
extern "C" void kernel_launch(void* const* d_in, const int* in_sizes, int n_in,
                              void* d_out, int out_size, void* d_ws, size_t ws_size,
                              hipStream_t stream) {
  (void)in_sizes; (void)n_in; (void)out_size; (void)ws_size;
  const float* x    = (const float*)d_in[0];
  const float* cyc  = (const float*)d_in[1];
  const float* dkp  = (const float*)d_in[2];
  const float* W1   = (const float*)d_in[3];
  const float* b1   = (const float*)d_in[4];
  const float* W2   = (const float*)d_in[5];
  const float* b2   = (const float*)d_in[6];
  const float* eW   = (const float*)d_in[7];
  const float* eB   = (const float*)d_in[8];
  const float* gW   = (const float*)d_in[9];
  const float* gB   = (const float*)d_in[10];
  float* out = (float*)d_out;

  float* h_ws = (float*)d_ws;                 // 64*2048 f32
  float* g_ws = h_ws + (size_t)BATT * DFF;    // 64*2 f32
  int*   i_ws = (int*)(g_ws + BATT * 2);      // 64*2 int

  router_h_kernel<<<dim3(DFF / 128), 256, 0, stream>>>(dkp, cyc, W1, b1, h_ws);
  router_gates_kernel<<<dim3(BATT), 128, 0, stream>>>(h_ws, W2, b2, g_ws, i_ws);
  moe_fused_gemm_kernel<<<dim3(DMODEL / 128, LCYC / 128, BATT), 256, 0, stream>>>(
      x, gW, gB, eW, eB, g_ws, i_ws, out);
}
